// Pyg_GAT_31104153158264
// MI455X (gfx1250) — matrix operand order repaired and checked by reading, not compiled, not run on hardware
//
#include <hip/hip_runtime.h>
#include <hip/hip_bf16.h>

// ---------------------------------------------------------------------------
// GAT forward for MI455X (gfx1250): WMMA f16 GEMMs + async-LDS B staging +
// L2-resident edge atomics.
// ---------------------------------------------------------------------------

typedef __attribute__((ext_vector_type(16))) _Float16 v16h;
typedef __attribute__((ext_vector_type(8)))  float    v8f;
typedef int v4i __attribute__((vector_size(16)));   // matches builtin param type

#define NN_NODES 50000
#define NF_FEATS 100
#define FEAT_PAD 128
#define NL_LABELS 19

#define AS1 __attribute__((address_space(1)))
#define AS3 __attribute__((address_space(3)))

#if defined(__has_builtin)
#if __has_builtin(__builtin_amdgcn_global_load_async_to_lds_b128) && \
    __has_builtin(__builtin_amdgcn_s_wait_asynccnt)
#define USE_ASYNC_LDS 1
#endif
#endif

// ---- helpers ---------------------------------------------------------------

__device__ __forceinline__ unsigned fkey(float f) {
    unsigned u = __float_as_uint(f);
    return (u & 0x80000000u) ? ~u : (u | 0x80000000u);   // order-preserving map
}
__device__ __forceinline__ float funkey(unsigned k) {
    return __uint_as_float((k & 0x80000000u) ? (k & 0x7fffffffu) : ~k);
}
__device__ __forceinline__ float lrelu(float v) { return v > 0.f ? v : 0.2f * v; }

// ---- WMMA GEMM: C[M x Nreal] = A_f16[M x Kpad] * B_f16[Kpad x Npad] --------
// B staged in LDS (async if available); each wave owns one 16-row M tile and
// loops over N tiles. Epilogue fully specialized at compile time.
template <int KSTEPS, bool BIAS, bool RELU, bool OUTF, bool OUTH>
__global__ void __launch_bounds__(256) gemm_wmma(
    const _Float16* __restrict__ A, int lda,
    const _Float16* __restrict__ B, int Npad,
    int Mtiles, int Ntiles, int Nreal,
    const float* __restrict__ bias,
    float* __restrict__ outF,
    _Float16* __restrict__ outH, int ldo)
{
    extern __shared__ __align__(16) char smem_raw[];
    _Float16* sB = (_Float16*)smem_raw;
    const int Kpad = KSTEPS * 32;

#if USE_ASYNC_LDS
    {   // async B-panel staging: GLOBAL_LOAD_ASYNC_TO_LDS_B128, ASYNCcnt-tracked
        const int chunks = (Kpad * Npad * 2) >> 4;          // 16B per lane-op
        const char* gb = (const char*)B;
        char*       sb = (char*)sB;
        for (int i = threadIdx.x; i < chunks; i += blockDim.x) {
            __builtin_amdgcn_global_load_async_to_lds_b128(
                (AS1 v4i*)(uintptr_t)(gb + ((size_t)i << 4)),
                (AS3 v4i*)(unsigned)(uintptr_t)(sb + ((size_t)i << 4)),
                0, 0);
        }
        __builtin_amdgcn_s_wait_asynccnt(0);
    }
#else
    {   // fallback: dword copy through VGPRs
        const unsigned* g = (const unsigned*)B;
        unsigned*       s = (unsigned*)sB;
        const int tot = (Kpad * Npad) >> 1;
        for (int i = threadIdx.x; i < tot; i += blockDim.x) s[i] = g[i];
    }
#endif
    __syncthreads();

    const int wave  = threadIdx.x >> 5;
    const int lane  = threadIdx.x & 31;
    const int mtile = blockIdx.x * 8 + wave;
    if (mtile >= Mtiles) return;

    const int mbase = mtile << 4;
    const int r     = lane & 15;
    const int koff  = (lane < 16) ? 0 : 8;
    const unsigned* arow = (const unsigned*)(A + (size_t)(mbase + r) * lda);

    // A fragments for all K steps (16-bit A 16x32 layout: lane<16 -> K 0..7,16..23;
    // lane>=16 -> K 8..15,24..31; packed pairs are K-contiguous -> dword loads)
    union AF { v16h h; unsigned u[8]; } af[KSTEPS];
#pragma unroll
    for (int ks = 0; ks < KSTEPS; ++ks) {
        const int kb = ks * 32;
#pragma unroll
        for (int v = 0; v < 4; ++v) {
            af[ks].u[v]     = arow[(kb + koff + 2 * v) >> 1];
            af[ks].u[4 + v] = arow[(kb + 16 + koff + 2 * v) >> 1];
        }
    }

    for (int nt = 0; nt < Ntiles; ++nt) {
        const int nb = nt << 4;
        v8f acc = {};
#pragma unroll
        for (int ks = 0; ks < KSTEPS; ++ks) {
            const int kb = ks * 32;
            // B 32x16 fragment: N = lane%16, K in same order as A (koff + s&7 + 16*(s>>3))
            union BF { v16h h; unsigned u[8]; } bf;
            const _Float16* bcol = sB + (size_t)kb * Npad + nb + r;
#pragma unroll
            for (int rr = 0; rr < 16; ++rr) bf.h[rr] = bcol[(size_t)(koff + (rr & 7) + ((rr >> 3) << 4)) * Npad];
            acc = __builtin_amdgcn_wmma_f32_16x16x32_f16(
                false, af[ks].h, false, bf.h, (short)0, acc, false, false);
        }
        // C/D layout: lane<16 -> N=lane, M=mbase+v ; lane>=16 -> N=lane-16, M=mbase+8+v
        const int n  = nb + r;
        const int m0 = mbase + ((lane < 16) ? 0 : 8);
        if (n < Nreal) {
            const float bs = BIAS ? bias[n] : 0.f;
#pragma unroll
            for (int v = 0; v < 8; ++v) {
                float val = acc[v] + bs;
                if (RELU) val = fmaxf(val, 0.f);
                const size_t m = (size_t)(m0 + v);
                if (OUTF) outF[m * (size_t)Nreal + n] = val;
                if (OUTH) outH[m * (size_t)ldo + n]  = (_Float16)val;
            }
        }
    }
}

// ---- small prep kernels ----------------------------------------------------

__global__ void clip_to_f16(const float* __restrict__ x, _Float16* __restrict__ xh,
                            int total) {
    const int i = blockIdx.x * blockDim.x + threadIdx.x;
    if (i >= total) return;
    const int n = i / NF_FEATS, f = i % NF_FEATS;
    const float v = fminf(fmaxf(x[i], -0.4f), 0.4f);
    xh[(size_t)n * FEAT_PAD + f] = (_Float16)v;
}

__global__ void convert_pad_w(const float* __restrict__ src, int K, int N,
                              _Float16* __restrict__ dst, int Kpad, int Npad) {
    const int i = blockIdx.x * blockDim.x + threadIdx.x;
    if (i >= Kpad * Npad) return;
    const int k = i / Npad, n = i % Npad;
    dst[i] = (k < K && n < N) ? (_Float16)src[(size_t)k * N + n] : (_Float16)0.f;
}

// ---- attention kernels -----------------------------------------------------

__global__ void node_dots(const float* __restrict__ H, const float* __restrict__ a_src,
                          const float* __restrict__ a_dst,
                          float* __restrict__ es, float* __restrict__ ed, int n_nodes) {
    const int wid  = (blockIdx.x * blockDim.x + threadIdx.x) >> 5;
    const int lane = threadIdx.x & 31;
    if (wid >= n_nodes) return;
    const float* row = H + (size_t)wid * NF_FEATS;
    float ps = 0.f, pd = 0.f;
    for (int f = lane; f < NF_FEATS; f += 32) {
        const float h = row[f];
        ps += h * a_src[f];
        pd += h * a_dst[f];
    }
#pragma unroll
    for (int o = 16; o > 0; o >>= 1) {
        ps += __shfl_xor(ps, o, 32);
        pd += __shfl_xor(pd, o, 32);
    }
    if (lane == 0) { es[wid] = ps; ed[wid] = pd; }
}

__global__ void edge_logit_max(const int* __restrict__ src, const int* __restrict__ dst,
                               const float* __restrict__ es, const float* __restrict__ ed,
                               float* __restrict__ le, float* __restrict__ ls,
                               unsigned* __restrict__ mkey, int E, int n_nodes) {
    const int idx = blockIdx.x * blockDim.x + threadIdx.x;
    if (idx < E) {
        const int j = src[idx], i = dst[idx];
        const float l = lrelu(es[j] + ed[i]);
        le[idx] = l;
        atomicMax(&mkey[i], fkey(l));
    } else if (idx < E + n_nodes) {
        const int n = idx - E;                      // self-loop
        const float l = lrelu(es[n] + ed[n]);
        ls[n] = l;
        atomicMax(&mkey[n], fkey(l));
    }
}

__global__ void edge_exp_sum(const int* __restrict__ dst,
                             float* __restrict__ le, float* __restrict__ ls,
                             const unsigned* __restrict__ mkey,
                             float* __restrict__ ssum, int E, int n_nodes) {
    const int idx = blockIdx.x * blockDim.x + threadIdx.x;
    if (idx < E) {
        const int i = dst[idx];
        const float p = expf(le[idx] - funkey(mkey[i]));
        le[idx] = p;
        atomicAdd(&ssum[i], p);
    } else if (idx < E + n_nodes) {
        const int n = idx - E;
        const float p = expf(ls[n] - funkey(mkey[n]));
        ls[n] = p;
        atomicAdd(&ssum[n], p);
    }
}

__global__ void edge_aggregate(const int* __restrict__ src, const int* __restrict__ dst,
                               const float* __restrict__ pe, const float* __restrict__ pn,
                               const float* __restrict__ ssum, const float* __restrict__ H,
                               float* __restrict__ Y, int E, int n_nodes) {
    const int wid  = (blockIdx.x * blockDim.x + threadIdx.x) >> 5;   // warp per edge
    const int lane = threadIdx.x & 31;
    if (wid >= E + n_nodes) return;
    int i, j; float p;
    if (wid < E) { j = src[wid]; i = dst[wid]; p = pe[wid]; }
    else         { i = j = wid - E;            p = pn[i]; }
    const float alpha = p / (ssum[i] + 1e-16f);
    const float* hr = H + (size_t)j * NF_FEATS;
    float*       yr = Y + (size_t)i * NF_FEATS;
    for (int f = lane; f < NF_FEATS; f += 32)
        atomicAdd(&yr[f], alpha * hr[f]);
}

__global__ void finalize_relu(const float* __restrict__ Y, const float* __restrict__ b,
                              _Float16* __restrict__ xout, int n_nodes) {
    const int i = blockIdx.x * blockDim.x + threadIdx.x;
    if (i >= n_nodes * NF_FEATS) return;
    const int n = i / NF_FEATS, f = i % NF_FEATS;
    const float v = fmaxf(Y[i] + b[f], 0.f);
    xout[(size_t)n * FEAT_PAD + f] = (_Float16)v;
}

__global__ void log_softmax_rows(const float* __restrict__ logits,
                                 float* __restrict__ out, int n_rows) {
    const int r = blockIdx.x * blockDim.x + threadIdx.x;
    if (r >= n_rows) return;
    const float* v = logits + (size_t)r * NL_LABELS;
    float m = -1e30f;
    for (int c = 0; c < NL_LABELS; ++c) m = fmaxf(m, v[c]);
    float s = 0.f;
    for (int c = 0; c < NL_LABELS; ++c) s += expf(v[c] - m);
    const float ls = logf(s);
    float* o = out + (size_t)r * NL_LABELS;
    for (int c = 0; c < NL_LABELS; ++c) o[c] = v[c] - m - ls;
}

// ---------------------------------------------------------------------------

extern "C" void kernel_launch(void* const* d_in, const int* in_sizes, int n_in,
                              void* d_out, int out_size, void* d_ws, size_t ws_size,
                              hipStream_t stream) {
    (void)n_in; (void)out_size; (void)ws_size;

    const float* x      = (const float*)d_in[0];
    const int*   ei     = (const int*)d_in[1];      // harness: integer -> int32
    const int    E      = in_sizes[1] / 2;
    const int*   esrc   = ei;                       // edge_index[0]
    const int*   edst   = ei + E;                   // edge_index[1]
    const float* W1     = (const float*)d_in[2];
    const float* a_src1 = (const float*)d_in[3];
    const float* a_dst1 = (const float*)d_in[4];
    const float* b1     = (const float*)d_in[5];
    const float* W2     = (const float*)d_in[6];
    const float* a_src2 = (const float*)d_in[7];
    const float* a_dst2 = (const float*)d_in[8];
    const float* b2     = (const float*)d_in[9];
    const float* fc1_w  = (const float*)d_in[10];
    const float* fc1_b  = (const float*)d_in[11];
    const float* fc2_w  = (const float*)d_in[12];
    const float* fc2_b  = (const float*)d_in[13];

    // ---- carve workspace ----
    char*  base = (char*)d_ws;
    size_t off  = 0;
    auto carve = [&](size_t bytes) -> void* {
        off = (off + 255) & ~(size_t)255;
        void* p = base + off;
        off += bytes;
        return p;
    };
    _Float16* xhA  = (_Float16*)carve((size_t)NN_NODES * FEAT_PAD * 2);
    _Float16* xhB  = (_Float16*)carve((size_t)NN_NODES * FEAT_PAD * 2);
    _Float16* xhC  = (_Float16*)carve((size_t)NN_NODES * 256 * 2);
    _Float16* W1p  = (_Float16*)carve((size_t)128 * 112 * 2);
    _Float16* W2p  = (_Float16*)carve((size_t)128 * 112 * 2);
    _Float16* F1p  = (_Float16*)carve((size_t)128 * 256 * 2);
    _Float16* F2p  = (_Float16*)carve((size_t)256 * 32 * 2);
    float*    H    = (float*)carve((size_t)NN_NODES * NF_FEATS * 4);
    float*    Y    = (float*)carve((size_t)NN_NODES * NF_FEATS * 4);
    float*    es   = (float*)carve((size_t)NN_NODES * 4);
    float*    ed   = (float*)carve((size_t)NN_NODES * 4);
    unsigned* mkey = (unsigned*)carve((size_t)NN_NODES * 4);
    float*    ssum = (float*)carve((size_t)NN_NODES * 4);
    float*    ls   = (float*)carve((size_t)NN_NODES * 4);
    float*    le   = (float*)carve((size_t)E * 4);
    float*    lg   = (float*)carve((size_t)NN_NODES * NL_LABELS * 4);

    const int Mtiles    = NN_NODES / 16;            // 3125, exact
    const int gemmGrid  = (Mtiles + 7) / 8;         // 8 waves / block
    const int totEN     = E + NN_NODES;

    // ---- prep: clip(x) -> f16 padded; convert all weights to padded f16 ----
    (void)hipMemsetAsync(xhA, 0, (size_t)NN_NODES * FEAT_PAD * 2, stream);
    clip_to_f16<<<(NN_NODES * NF_FEATS + 255) / 256, 256, 0, stream>>>(
        x, xhA, NN_NODES * NF_FEATS);
    convert_pad_w<<<(128 * 112 + 255) / 256, 256, 0, stream>>>(W1, 100, 100, W1p, 128, 112);
    convert_pad_w<<<(128 * 112 + 255) / 256, 256, 0, stream>>>(W2, 100, 100, W2p, 128, 112);
    convert_pad_w<<<(128 * 256 + 255) / 256, 256, 0, stream>>>(fc1_w, 100, 256, F1p, 128, 256);
    convert_pad_w<<<(256 * 32 + 255) / 256, 256, 0, stream>>>(fc2_w, 256, 19, F2p, 256, 32);

    // ---- one GAT layer ----
    auto gat_layer = [&](const _Float16* xin, const _Float16* Wp, const float* as,
                         const float* ad, const float* b, _Float16* xout) {
        // H = xin @ W  (f16 WMMA, f32 out, no bias/relu)
        gemm_wmma<4, false, false, true, false><<<gemmGrid, 256, 128 * 112 * 2, stream>>>(
            xin, 128, Wp, 112, Mtiles, 7, 100,
            nullptr, H, (_Float16*)nullptr, 0);
        node_dots<<<(NN_NODES * 32 + 255) / 256, 256, 0, stream>>>(H, as, ad, es, ed, NN_NODES);
        (void)hipMemsetAsync(mkey, 0, (size_t)NN_NODES * 4, stream);   // key(0)=NaN-floor; self-loop guarantees a real max
        (void)hipMemsetAsync(ssum, 0, (size_t)NN_NODES * 4, stream);
        (void)hipMemsetAsync(Y, 0, (size_t)NN_NODES * NF_FEATS * 4, stream);
        edge_logit_max<<<(totEN + 255) / 256, 256, 0, stream>>>(
            esrc, edst, es, ed, le, ls, mkey, E, NN_NODES);
        edge_exp_sum<<<(totEN + 255) / 256, 256, 0, stream>>>(
            edst, le, ls, mkey, ssum, E, NN_NODES);
        edge_aggregate<<<(int)(((long long)totEN * 32 + 255) / 256), 256, 0, stream>>>(
            esrc, edst, le, ls, ssum, H, Y, E, NN_NODES);
        (void)hipMemsetAsync(xout, 0, (size_t)NN_NODES * FEAT_PAD * 2, stream);
        finalize_relu<<<(NN_NODES * NF_FEATS + 255) / 256, 256, 0, stream>>>(
            Y, b, xout, NN_NODES);
    };

    gat_layer(xhA, W1p, a_src1, a_dst1, b1, xhB);   // layer 1: xhA -> xhB
    gat_layer(xhB, W2p, a_src2, a_dst2, b2, xhA);   // layer 2: xhB -> xhA (reuse)

    // fc1: relu(x @ fc1_w + b), f16 out for fc2
    gemm_wmma<4, true, true, false, true><<<gemmGrid, 256, 128 * 256 * 2, stream>>>(
        xhA, 128, F1p, 256, Mtiles, 16, 256,
        fc1_b, (float*)nullptr, xhC, 256);
    // fc2: logits = x @ fc2_w + b (f32 out)
    gemm_wmma<8, true, false, true, false><<<gemmGrid, 256, 256 * 32 * 2, stream>>>(
        xhC, 256, F2p, 32, Mtiles, 2, 19,
        fc2_b, lg, (_Float16*)nullptr, 0);

    log_softmax_rows<<<(NN_NODES + 255) / 256, 256, 0, stream>>>(
        lg, (float*)d_out, NN_NODES);
}
